// WordLSTM_86990267613907
// MI455X (gfx1250) — matrix operand order repaired and checked by reading, not compiled, not run on hardware
//
#include <hip/hip_runtime.h>

typedef _Float16 half_t;
typedef _Float16 v16h __attribute__((ext_vector_type(16)));
typedef float    v8f  __attribute__((ext_vector_type(8)));

#define BATCH    8192
#define T_SEQ    200
#define EMB_D    20
#define HID_D    20
#define NLAB     15
#define GATES    80      // 4*HID
#define HX_STRIDE 72     // halves per row of A-staging ([h|e] padded to K=64, +pad)
#define Z_STRIDE  84     // floats per row of z (8*84 % 64 == 32 -> conflict-free halves)
#define WT_STRIDE 72     // halves per column of transposed weight staging
#define WAVES    4
#define ROWS_PER_BLOCK (WAVES * 16)

union AFragU { v16h v; struct { int4 lo; int4 hi; } s; };

// CDNA5 has V_TANH_F32 (TRANS32). Use it for all gate nonlinearities:
//   tanh(x) directly; sigmoid(x) = 0.5*tanh(0.5*x) + 0.5
#if __has_builtin(__builtin_amdgcn_tanhf)
__device__ __forceinline__ float fast_tanh(float x) { return __builtin_amdgcn_tanhf(x); }
#else
__device__ __forceinline__ float fast_tanh(float x) {
    return 2.f / (1.f + __expf(-2.f * x)) - 1.f;
}
#endif
__device__ __forceinline__ float fast_sigmoid(float x) {
    return fmaf(0.5f, fast_tanh(0.5f * x), 0.5f);
}

__global__ __launch_bounds__(128) void lstm_fused(
    const int*   __restrict__ x,    // [B, T]
    const float* __restrict__ emb,  // [VOCAB, 20]
    const float* __restrict__ Wx,   // [20, 80]
    const float* __restrict__ Wh,   // [20, 80]
    const float* __restrict__ bias, // [80]
    const float* __restrict__ Wd,   // [20, 15]
    const float* __restrict__ bd,   // [15]
    float*       __restrict__ out)  // [B, 15]
{
    // Combined weight, transposed + gate-interleaved: WT[col][k], col = 4*j + p
    // (p = gate i/f/g/o, j = hidden index), k<20 -> Wh row k, 32<=k<52 -> Wx row k-32.
    __shared__ half_t WT[GATES][WT_STRIDE];
    __shared__ float  WdS[HID_D * NLAB];
    __shared__ float  bdS[NLAB];
    __shared__ half_t HX[WAVES][2][16][HX_STRIDE]; // A staging: [h(K0..19) | e(K32..51)], f16
    __shared__ float  Z[WAVES][16][Z_STRIDE];      // gate pre-activations

    const int tid  = threadIdx.x;
    const int wave = tid >> 5;
    const int lane = tid & 31;
    const int hb   = lane >> 4;   // wave half
    const int ln   = lane & 15;

    // ---------------- one-time staging ----------------
    for (int idx = tid; idx < GATES * WT_STRIDE; idx += 128) {
        int col = idx / WT_STRIDE, k = idx - col * WT_STRIDE;
        int j = col >> 2, p = col & 3, oc = p * HID_D + j;   // original column
        float v = 0.f;
        if (k < HID_D)                    v = Wh[k * GATES + oc];
        else if (k >= 32 && k < 32+EMB_D) v = Wx[(k - 32) * GATES + oc];
        WT[col][k] = (half_t)v;
    }
    for (int idx = tid; idx < HID_D * NLAB; idx += 128) WdS[idx] = Wd[idx];
    if (tid < NLAB) bdS[tid] = bd[tid];
    {   // zero A-staging (covers h-init=0 and all K padding)
        half_t* hp = &HX[wave][0][0][0];
        for (int i = lane; i < 2 * 16 * HX_STRIDE; i += 32) hp[i] = (half_t)0.f;
    }
    __syncthreads();

    // ---------------- B fragments (weights) held in registers ----------------
    v16h Bf[5][2];
    #pragma unroll
    for (int nt = 0; nt < 5; ++nt)
        #pragma unroll
        for (int kc = 0; kc < 2; ++kc) {
            // lane = column nt*16+ln; lanes 0-15 hold K=kc*32+0..15, lanes 16-31 K=+16..31
            const int4* p = (const int4*)&WT[nt*16 + ln][kc*32 + hb*16];
            AFragU u; u.s.lo = p[0]; u.s.hi = p[1];
            Bf[nt][kc] = u.v;
        }
    float biasv[5];
    #pragma unroll
    for (int nt = 0; nt < 5; ++nt) {
        int col = nt*16 + ln;
        biasv[nt] = bias[(col & 3) * HID_D + (col >> 2)];
    }

    const int rowb = blockIdx.x * ROWS_PER_BLOCK + wave * 16;
    float creg[10], hreg[10];
    #pragma unroll
    for (int q = 0; q < 10; ++q) { creg[q] = 0.f; hreg[q] = 0.f; }

    // embedding gather mapping: 2 lanes per row, 10 floats each
    const int grow = lane >> 1;
    const int goff = (lane & 1) * 10;

    int tok_cur = 0, tok_nxt = 0;
    if (lane < 16) {
        tok_cur = x[(rowb + lane) * T_SEQ + 0];
        tok_nxt = x[(rowb + lane) * T_SEQ + 1];
    }
    // gather e[0] into buffer 0 (f32 -> f16 pairs, b32 LDS stores)
    {
        int tr = __shfl(tok_cur, grow);
        const float* ep = emb + tr * EMB_D + goff;
        #pragma unroll
        for (int k2 = 0; k2 < 5; ++k2) {
            float2 v = *(const float2*)(ep + 2*k2);
            half_t h0 = (half_t)v.x, h1 = (half_t)v.y;
            unsigned u = (unsigned)__builtin_bit_cast(unsigned short, h0)
                       | ((unsigned)__builtin_bit_cast(unsigned short, h1) << 16);
            *(unsigned*)&HX[wave][0][grow][32 + goff + 2*k2] = u;
        }
    }

    // ---------------- recurrence ----------------
    int buf = 0;
    for (int t = 0; t < T_SEQ; ++t) {
        const int nb = buf ^ 1;
        // prefetch e[t+1] into the alternate buffer while this step computes
        if (t + 1 < T_SEQ) {
            int tr = __shfl(tok_nxt, grow);
            const float* ep = emb + tr * EMB_D + goff;
            #pragma unroll
            for (int k2 = 0; k2 < 5; ++k2) {
                float2 v = *(const float2*)(ep + 2*k2);
                half_t h0 = (half_t)v.x, h1 = (half_t)v.y;
                unsigned u = (unsigned)__builtin_bit_cast(unsigned short, h0)
                           | ((unsigned)__builtin_bit_cast(unsigned short, h1) << 16);
                *(unsigned*)&HX[wave][nb][grow][32 + goff + 2*k2] = u;
            }
        }
        int tok_n2 = 0;
        if (lane < 16 && t + 2 < T_SEQ) tok_n2 = x[(rowb + lane) * T_SEQ + t + 2];

        // A fragments: row m=ln; lanes0-15: K {0..15}, lanes16-31: K {16..31}
        v16h Af[2];
        #pragma unroll
        for (int kc = 0; kc < 2; ++kc) {
            const int4* p = (const int4*)&HX[wave][buf][ln][kc*32 + hb*16];
            AFragU u; u.s.lo = p[0]; u.s.hi = p[1];   // +0 and +8 halves
            Af[kc] = u.v;
        }

        // z = [h|e] @ W + b  (bias folded into accumulator init)
        v8f acc[5];
        #pragma unroll
        for (int nt = 0; nt < 5; ++nt) {
            #pragma unroll
            for (int r = 0; r < 8; ++r) acc[nt][r] = biasv[nt];
        }
        #pragma unroll
        for (int nt = 0; nt < 5; ++nt)
            #pragma unroll
            for (int kc = 0; kc < 2; ++kc)
                acc[nt] = __builtin_amdgcn_wmma_f32_16x16x32_f16(
                    false, Af[kc], false, Bf[nt][kc], (short)0, acc[nt], false, false);

        // spill z to LDS (C layout: row = r + 8*hb, col = nt*16 + ln)
        #pragma unroll
        for (int nt = 0; nt < 5; ++nt)
            #pragma unroll
            for (int r = 0; r < 8; ++r)
                Z[wave][r + 8*hb][nt*16 + ln] = acc[nt][r];

        // elementwise gate math: each lane owns 10 of the 320 (m,j) cells;
        // gate-interleaved columns make i,f,g,o one float4 read.
        #pragma unroll
        for (int q = 0; q < 10; ++q) {
            int idx = lane + 32*q;
            int m = idx / 20, j = idx - m*20;
            float4 g4 = *(const float4*)&Z[wave][m][4*j];
            float ig = fast_sigmoid(g4.x);
            float fg = fast_sigmoid(g4.y);
            float gg = fast_tanh(g4.z);
            float og = fast_sigmoid(g4.w);
            float c  = fmaf(fg, creg[q], ig * gg);
            creg[q]  = c;
            float h  = og * fast_tanh(c);
            hreg[q]  = h;
            HX[wave][nb][m][j] = (half_t)h;   // next step's A (h region)
        }
        tok_nxt = tok_n2;
        buf = nb;
    }

    // ---------------- dense head + softmax ----------------
    #pragma unroll
    for (int q = 0; q < 10; ++q) {
        int idx = lane + 32*q;
        int m = idx / 20, j = idx - m*20;
        Z[wave][m][j] = hreg[q];             // final h, f32, reuse Z
    }
    if (lane < 16) {
        const int m = lane;
        float hv[HID_D];
        #pragma unroll
        for (int j = 0; j < HID_D; ++j) hv[j] = Z[wave][m][j];
        float logits[NLAB];
        float mx = -1e30f;
        #pragma unroll
        for (int l = 0; l < NLAB; ++l) {
            float s = bdS[l];
            #pragma unroll
            for (int j = 0; j < HID_D; ++j) s += hv[j] * WdS[j * NLAB + l];
            logits[l] = s;
            mx = fmaxf(mx, s);
        }
        float sum = 0.f;
        #pragma unroll
        for (int l = 0; l < NLAB; ++l) { float e = __expf(logits[l] - mx); logits[l] = e; sum += e; }
        float inv = 1.f / sum;
        float* op = out + (rowb + m) * NLAB;
        #pragma unroll
        for (int l = 0; l < NLAB; ++l) op[l] = logits[l] * inv;
    }
}

extern "C" void kernel_launch(void* const* d_in, const int* in_sizes, int n_in,
                              void* d_out, int out_size, void* d_ws, size_t ws_size,
                              hipStream_t stream) {
    const int*   x   = (const int*)  d_in[0];
    const float* emb = (const float*)d_in[1];
    const float* Wx  = (const float*)d_in[2];
    const float* Wh  = (const float*)d_in[3];
    const float* b   = (const float*)d_in[4];
    const float* Wd  = (const float*)d_in[5];
    const float* bd  = (const float*)d_in[6];
    float* out = (float*)d_out;
    (void)in_sizes; (void)n_in; (void)out_size; (void)d_ws; (void)ws_size;

    dim3 grid(BATCH / ROWS_PER_BLOCK);   // 128 blocks
    dim3 block(128);                     // 4 wave32 per block, one 16-row tile each
    hipLaunchKernelGGL(lstm_fused, grid, block, 0, stream,
                       x, emb, Wx, Wh, b, Wd, bd, out);
}
